// TransformerEncoderBlock_46720654246412
// MI455X (gfx1250) — matrix operand order repaired and checked by reading, not compiled, not run on hardware
//
#include <hip/hip_runtime.h>
#include <stdint.h>

// ---------------------------------------------------------------------------
// MI455X / gfx1250 transformer encoder block.
// All GEMMs via v_wmma_f32_16x16x32_bf16 (wave32, 16x16 tiles, f32 accum).
// ~155 GFLOP total vs ~0.2 GB HBM traffic -> compute bound -> bf16 matrix path.
// Attention is flash-style (online softmax), never materializing S x S scores.
// GEMM tiles double-buffered through LDS using gfx1250 async global->LDS DMA
// (ASYNCcnt) when the toolchain exposes the builtins; sync copy otherwise.
// ---------------------------------------------------------------------------

typedef __attribute__((ext_vector_type(16))) __bf16 v16bf;
typedef __attribute__((ext_vector_type(8)))  float  v8f;
typedef int vsi4 __attribute__((vector_size(16)));   // matches async-LDS builtin

#if defined(__HIP_DEVICE_COMPILE__) && defined(__gfx1250__) && \
    __has_builtin(__builtin_amdgcn_global_load_async_to_lds_b128) && \
    __has_builtin(__builtin_amdgcn_s_wait_asynccnt)
#define USE_ASYNC_LDS 1
#else
#define USE_ASYNC_LDS 0
#endif

union Frag {
  v16bf  v;
  uint4  q[2];
  __bf16 h[16];
};
union Chunk8 {
  uint4  q;
  __bf16 h[8];
};

static __device__ __forceinline__ v8f vzero8() {
  v8f z = {0.f, 0.f, 0.f, 0.f, 0.f, 0.f, 0.f, 0.f};
  return z;
}

static __device__ __forceinline__ v8f wmma_bf16(const Frag& a, const Frag& b, v8f c) {
  // D = A(16x32 bf16) * B(32x16 bf16) + C(16x16 f32)
  return __builtin_amdgcn_wmma_f32_16x16x32_bf16(false, a.v, false, b.v,
                                                 (short)0, c, false, false);
}

#if USE_ASYNC_LDS
// 16-byte async global -> LDS copy (per-lane), tracked by ASYNCcnt.
static __device__ __forceinline__ void async_cp16(const __bf16* g, __bf16* l) {
  __builtin_amdgcn_global_load_async_to_lds_b128(
      (__attribute__((address_space(1))) vsi4*)const_cast<__bf16*>(g),
      (__attribute__((address_space(3))) vsi4*)l, 0, 0);
}
#endif

// ---------------------------------------------------------------------------
// Elementwise f32 -> bf16
// ---------------------------------------------------------------------------
__global__ __launch_bounds__(256) void cvt_kernel(const float* __restrict__ in,
                                                  __bf16* __restrict__ out, int n) {
  int i = blockIdx.x * 256 + threadIdx.x;
  if (i < n) out[i] = (__bf16)in[i];
}

// Transpose-convert: in[R][C] f32 -> out[C][R] bf16. blockIdx.y selects matrix slab.
__global__ __launch_bounds__(256) void cvtT_kernel(const float* __restrict__ in,
                                                   __bf16* __restrict__ out,
                                                   int R, int C) {
  in  += (size_t)blockIdx.y * R * C;
  out += (size_t)blockIdx.y * R * C;
  int i = blockIdx.x * 256 + threadIdx.x;
  if (i < R * C) {
    int r = i / C, c = i - r * C;
    out[(size_t)c * R + r] = (__bf16)in[i];
  }
}

// ---------------------------------------------------------------------------
// Tiled bf16 WMMA GEMM.  A: [M][K] bf16 row-major.  Bt: [N][K] bf16 (B^T).
// Block tile BM x BN, K-step 32, 8 waves arranged WR x WC, double-buffered LDS.
// EPI: 0 = relu(acc+bias) -> bf16 ; 1 = acc+bias+resid -> f32
//      3 = per-head q-projection, scatter to qh[b][h][s][dh] (h = blockIdx.z)
// ---------------------------------------------------------------------------
template <int BM, int BN, int WR, int WC, int EPI>
__global__ __launch_bounds__(256)
void gemm_kernel(const __bf16* __restrict__ A, const __bf16* __restrict__ Bt,
                 const float* __restrict__ bias, const float* __restrict__ resid,
                 float* __restrict__ outF, __bf16* __restrict__ outB,
                 int M, int N, int K) {
  constexpr int BK  = 32;
  constexpr int LR  = 40;              // LDS row stride (32 data + 8 pad halves)
  constexpr int WTM = BM / WR, WTN = BN / WC;
  constexpr int MT  = WTM / 16, NT = WTN / 16;

  __shared__ __align__(16) __bf16 lds[2][(BM + BN) * LR];

  const int tid  = threadIdx.x;
  const int lane = tid & 31;
  const int wid  = tid >> 5;
  const int wr   = wid % WR;
  const int wc   = wid / WR;
  const int lm   = lane & 15;
  const int g    = lane >> 4;

  const int bm0 = blockIdx.y * BM;
  const int bn0 = blockIdx.x * BN;

  if constexpr (EPI == 3) {
    Bt   += (size_t)blockIdx.z * 64 * K;   // head slab of wqT: [Dh][K]
    bias += blockIdx.z * 64;               // wq_b[h]
  }

  v8f acc[MT][NT];
#pragma unroll
  for (int i = 0; i < MT; ++i)
#pragma unroll
    for (int j = 0; j < NT; ++j) acc[i][j] = vzero8();

  // Stage one BK-slab of A and Bt into LDS buffer p.
  auto stage = [&](int k0, int p) {
    __bf16* dA = lds[p];
    __bf16* dB = lds[p] + BM * LR;
#if USE_ASYNC_LDS
#pragma unroll
    for (int c = tid; c < BM * 4; c += 256) {
      int r = c >> 2, cj = c & 3;
      async_cp16(A + (size_t)(bm0 + r) * K + k0 + cj * 8, dA + r * LR + cj * 8);
    }
#pragma unroll
    for (int c = tid; c < BN * 4; c += 256) {
      int r = c >> 2, cj = c & 3;
      async_cp16(Bt + (size_t)(bn0 + r) * K + k0 + cj * 8, dB + r * LR + cj * 8);
    }
#else
#pragma unroll
    for (int c = tid; c < BM * 4; c += 256) {
      int r = c >> 2, cj = c & 3;
      *(uint4*)(dA + r * LR + cj * 8) =
          *(const uint4*)(A + (size_t)(bm0 + r) * K + k0 + cj * 8);
    }
#pragma unroll
    for (int c = tid; c < BN * 4; c += 256) {
      int r = c >> 2, cj = c & 3;
      *(uint4*)(dB + r * LR + cj * 8) =
          *(const uint4*)(Bt + (size_t)(bn0 + r) * K + k0 + cj * 8);
    }
    if (k0 + BK < K) {   // gfx1250 global_prefetch for the slab after next
      __builtin_prefetch(A + (size_t)(bm0 + (tid & (BM - 1))) * K + k0 + BK, 0, 0);
      __builtin_prefetch(Bt + (size_t)(bn0 + (tid & (BN - 1))) * K + k0 + BK, 0, 0);
    }
#endif
  };

  stage(0, 0);
#if USE_ASYNC_LDS
  __builtin_amdgcn_s_wait_asynccnt(0);
#endif
  __syncthreads();

  for (int k0 = 0; k0 < K; k0 += BK) {
    const int p = (k0 / BK) & 1;
    if (k0 + BK < K) stage(k0 + BK, p ^ 1);   // overlap next slab with math

    const uint4* lA4 = (const uint4*)(lds[p]);            // row stride = 5 uint4
    const uint4* lB4 = (const uint4*)(lds[p] + BM * LR);

    Frag af[MT], bfr[NT];
#pragma unroll
    for (int mt = 0; mt < MT; ++mt) {
      int r = wr * WTM + mt * 16 + lm;       // A frag: lane = M row
      af[mt].q[0] = lA4[r * 5 + g];          // K = g*8 .. g*8+7
      af[mt].q[1] = lA4[r * 5 + g + 2];      // K = 16+g*8 .. 16+g*8+7
    }
#pragma unroll
    for (int nt = 0; nt < NT; ++nt) {
      int r = wc * WTN + nt * 16 + lm;       // B frag: lane = N col (row of Bt)
      bfr[nt].q[0] = lB4[r * 5 + g];         // K = g*8 .. g*8+7
      bfr[nt].q[1] = lB4[r * 5 + g + 2];     // K = 16+g*8 .. 16+g*8+7
    }
#pragma unroll
    for (int mt = 0; mt < MT; ++mt)
#pragma unroll
      for (int nt = 0; nt < NT; ++nt)
        acc[mt][nt] = wmma_bf16(af[mt], bfr[nt], acc[mt][nt]);

#if USE_ASYNC_LDS
    __builtin_amdgcn_s_wait_asynccnt(0);     // next slab landed in LDS
#endif
    __syncthreads();                         // all waves done reading slab p
  }

#pragma unroll
  for (int mt = 0; mt < MT; ++mt) {
#pragma unroll
    for (int nt = 0; nt < NT; ++nt) {
#pragma unroll
      for (int r = 0; r < 8; ++r) {
        int row = bm0 + wr * WTM + mt * 16 + r + 8 * g;  // D layout: M = r + 8*g
        int col = bn0 + wc * WTN + nt * 16 + lm;         //           N = lane%16
        float v = acc[mt][nt][r] + bias[col];
        if constexpr (EPI == 0) {
          outB[(size_t)row * N + col] = (__bf16)fmaxf(v, 0.0f);
        } else if constexpr (EPI == 1) {
          outF[(size_t)row * N + col] = v + resid[(size_t)row * N + col];
        } else {  // EPI==3: qh[b][h][s][dh], b=row/2048, s=row%2048, h=blockIdx.z
          size_t o = ((size_t)(row >> 11) * 16 + blockIdx.z) * (2048ull * 64) +
                     (size_t)(row & 2047) * 64 + col;
          outB[o] = (__bf16)v;
        }
      }
    }
  }
}

// ---------------------------------------------------------------------------
// Flash attention (Q=K=V=qh per reference bug). grid = (S/64, B*H), block 128.
// Each wave owns 16 query rows; keys streamed in chunks of 32 through LDS.
// ---------------------------------------------------------------------------
__global__ __launch_bounds__(128)
void attn_kernel(const __bf16* __restrict__ qh, __bf16* __restrict__ yout) {
  constexpr int S = 2048, DH = 64;
  const int bh = blockIdx.y;                 // b*16 + h
  const __bf16* Q = qh + (size_t)bh * S * DH;

  const int tid = threadIdx.x;
  const int lane = tid & 31;
  const int w = tid >> 5;                    // wave 0..3
  const int lm = lane & 15;
  const int g = lane >> 4;
  const int qrow0 = blockIdx.x * 64 + w * 16;

  __shared__ __align__(16) __bf16 kv[32 * DH];     // [key][dh]
  __shared__ __align__(16) __bf16 kvT[DH * 40];    // [dh][key], padded row
  __shared__ __align__(16) __bf16 pl[4][16 * 40];  // per-wave P tile, padded

  // Q fragments (A layout), kept in registers for the whole kernel.
  Frag qf[2];
  {
    const uint4* Qr = (const uint4*)(Q + (size_t)(qrow0 + lm) * DH);
#pragma unroll
    for (int ks = 0; ks < 2; ++ks) {
      qf[ks].q[0] = Qr[ks * 4 + g];
      qf[ks].q[1] = Qr[ks * 4 + g + 2];
    }
  }

  v8f ctx[4];
#pragma unroll
  for (int nt = 0; nt < 4; ++nt) ctx[nt] = vzero8();
  float mrow[8], lrow[8];
#pragma unroll
  for (int r = 0; r < 8; ++r) { mrow[r] = -1e30f; lrow[r] = 0.f; }

  const uint4* kvT4 = (const uint4*)kvT;
  const uint4* pw4  = (const uint4*)(pl[w]);

  for (int kc = 0; kc < S / 32; ++kc) {
    __syncthreads();
    // Stage 32 keys x 64 dh (4 KB) + transposed copy (register-based transpose).
#pragma unroll
    for (int j = 0; j < 2; ++j) {
      int c = tid * 2 + j;          // 0..255 chunks of 8 halves
      int key = c >> 3;
      int dh0 = (c & 7) * 8;
      Chunk8 ch;
      ch.q = *(const uint4*)(Q + (size_t)(kc * 32 + key) * DH + dh0);
      *(uint4*)(kv + key * DH + dh0) = ch.q;
#pragma unroll
      for (int i = 0; i < 8; ++i) kvT[(dh0 + i) * 40 + key] = ch.h[i];
    }
    __syncthreads();

    // scores: two 16-key column tiles, K = DH = 64 -> 2 wmma steps each
    v8f s[2];
#pragma unroll
    for (int kt = 0; kt < 2; ++kt) {
      s[kt] = vzero8();
      const uint4* kr = (const uint4*)(kv + (kt * 16 + lm) * DH);
#pragma unroll
      for (int ks = 0; ks < 2; ++ks) {
        Frag kf;                               // B frag: lane = key col
        kf.q[0] = kr[ks * 4 + g];
        kf.q[1] = kr[ks * 4 + g + 2];
        s[kt] = wmma_bf16(qf[ks], kf, s[kt]);
      }
    }

    // online softmax update (row = r + 8*g, cols across 16 lanes of a group)
#pragma unroll
    for (int r = 0; r < 8; ++r) {
      float a = s[0][r] * 0.125f;              // 1/sqrt(64)
      float b = s[1][r] * 0.125f;
      float t = fmaxf(a, b);
      t = fmaxf(t, __shfl_xor(t, 1));
      t = fmaxf(t, __shfl_xor(t, 2));
      t = fmaxf(t, __shfl_xor(t, 4));
      t = fmaxf(t, __shfl_xor(t, 8));
      float mn = fmaxf(mrow[r], t);
      float al = __expf(mrow[r] - mn);
      float p0 = __expf(a - mn);
      float p1 = __expf(b - mn);
      float su = p0 + p1;
      su += __shfl_xor(su, 1);
      su += __shfl_xor(su, 2);
      su += __shfl_xor(su, 4);
      su += __shfl_xor(su, 8);
      lrow[r] = lrow[r] * al + su;
      mrow[r] = mn;
#pragma unroll
      for (int nt = 0; nt < 4; ++nt) ctx[nt][r] *= al;
      pl[w][(r + 8 * g) * 40 + lm]      = (__bf16)p0;   // D-layout -> LDS
      pl[w][(r + 8 * g) * 40 + 16 + lm] = (__bf16)p1;
    }
    __syncthreads();

    // ctx += P(16x32) @ V(32x64)
    Frag pa;                                   // A frag from per-wave P tile
    pa.q[0] = pw4[lm * 5 + g];
    pa.q[1] = pw4[lm * 5 + g + 2];
#pragma unroll
    for (int nt = 0; nt < 4; ++nt) {
      Frag vf;                                 // B frag: lane = dh col, K = keys
      int dh = nt * 16 + lm;
      vf.q[0] = kvT4[dh * 5 + g];
      vf.q[1] = kvT4[dh * 5 + g + 2];
      ctx[nt] = wmma_bf16(pa, vf, ctx[nt]);
    }
  }

  // normalize and write concat-head output [B][S][D] as bf16
  const int b = bh >> 4, h = bh & 15;
#pragma unroll
  for (int nt = 0; nt < 4; ++nt) {
#pragma unroll
    for (int r = 0; r < 8; ++r) {
      int srow = qrow0 + r + 8 * g;
      int col = h * 64 + nt * 16 + lm;
      yout[((size_t)(b * S + srow)) * 1024 + col] = (__bf16)(ctx[nt][r] / lrow[r]);
    }
  }
}

// ---------------------------------------------------------------------------
// LayerNorm over D=1024, one block per row. Emits f32 (+ optional bf16 copy).
// ---------------------------------------------------------------------------
__global__ __launch_bounds__(256)
void ln_kernel(const float* __restrict__ z, const float* __restrict__ gam,
               const float* __restrict__ bet, float* __restrict__ outF,
               __bf16* __restrict__ outB) {
  const int row = blockIdx.x;
  const int tid = threadIdx.x;
  const float4 v = ((const float4*)(z + (size_t)row * 1024))[tid];

  float s  = v.x + v.y + v.z + v.w;
  float ss = v.x * v.x + v.y * v.y + v.z * v.z + v.w * v.w;
#pragma unroll
  for (int off = 1; off < 32; off <<= 1) {
    s  += __shfl_xor(s, off);
    ss += __shfl_xor(ss, off);
  }
  __shared__ float red[16];
  const int wid = tid >> 5;
  if ((tid & 31) == 0) { red[wid] = s; red[wid + 8] = ss; }
  __syncthreads();
  float st = 0.f, sst = 0.f;
#pragma unroll
  for (int i = 0; i < 8; ++i) { st += red[i]; sst += red[i + 8]; }

  const float mu   = st * (1.0f / 1024.0f);
  const float var  = sst * (1.0f / 1024.0f) - mu * mu;
  const float rinv = rsqrtf(var + 1e-5f);

  const float4 gv = ((const float4*)gam)[tid];
  const float4 bv = ((const float4*)bet)[tid];
  float4 o;
  o.x = (v.x - mu) * rinv * gv.x + bv.x;
  o.y = (v.y - mu) * rinv * gv.y + bv.y;
  o.z = (v.z - mu) * rinv * gv.z + bv.z;
  o.w = (v.w - mu) * rinv * gv.w + bv.w;
  ((float4*)(outF + (size_t)row * 1024))[tid] = o;
  if (outB) {
    size_t base = (size_t)row * 1024 + tid * 4;
    outB[base + 0] = (__bf16)o.x;
    outB[base + 1] = (__bf16)o.y;
    outB[base + 2] = (__bf16)o.z;
    outB[base + 3] = (__bf16)o.w;
  }
}

// ---------------------------------------------------------------------------
// Orchestration
// ---------------------------------------------------------------------------
extern "C" void kernel_launch(void* const* d_in, const int* in_sizes, int n_in,
                              void* d_out, int out_size, void* d_ws, size_t ws_size,
                              hipStream_t stream) {
  (void)in_sizes; (void)n_in; (void)out_size; (void)ws_size;
  const float* x    = (const float*)d_in[0];   // [2,2048,1024]
  const float* wq_w = (const float*)d_in[1];   // [16,1024,64]
  const float* wq_b = (const float*)d_in[2];   // [16,64]
  const float* fc_w = (const float*)d_in[3];   // [1024,1024]
  const float* fc_b = (const float*)d_in[4];
  const float* ln1g = (const float*)d_in[5];
  const float* ln1b = (const float*)d_in[6];
  const float* w1   = (const float*)d_in[7];   // [1024,4096]
  const float* b1   = (const float*)d_in[8];
  const float* w2   = (const float*)d_in[9];   // [4096,1024]
  const float* b2   = (const float*)d_in[10];
  const float* ln2g = (const float*)d_in[11];
  const float* ln2b = (const float*)d_in[12];

  char* ws = (char*)d_ws;
  size_t off = 0;
  auto take = [&](size_t bytes) -> char* {
    char* p = ws + off;
    off += (bytes + 255) & ~(size_t)255;
    return p;
  };
  __bf16* xb   = (__bf16*)take(4096ull * 1024 * 2);  // x as bf16
  __bf16* wqT  = (__bf16*)take(16ull * 64 * 1024 * 2);
  __bf16* fcT  = (__bf16*)take(1024ull * 1024 * 2);
  __bf16* w1T  = (__bf16*)take(4096ull * 1024 * 2);
  __bf16* w2T  = (__bf16*)take(1024ull * 4096 * 2);
  __bf16* qhB  = (__bf16*)take(4096ull * 1024 * 2);  // [B][H][S][Dh]
  __bf16* ya   = (__bf16*)take(4096ull * 1024 * 2);  // attention output
  float*  z1   = (float*)take(4096ull * 1024 * 4);   // pre-LN buffer (reused)
  float*  ylnF = (float*)take(4096ull * 1024 * 4);
  __bf16* ylnB = (__bf16*)take(4096ull * 1024 * 2);
  __bf16* h1   = (__bf16*)take(4096ull * 4096 * 2);  // MLP hidden

  // precision prep: one-shot bf16 (and transposed-bf16 for B operands)
  cvt_kernel<<<dim3(4096 * 1024 / 256), 256, 0, stream>>>(x, xb, 4096 * 1024);
  cvtT_kernel<<<dim3(1024 * 64 / 256, 16), 256, 0, stream>>>(wq_w, wqT, 1024, 64);
  cvtT_kernel<<<dim3(1024 * 1024 / 256, 1), 256, 0, stream>>>(fc_w, fcT, 1024, 1024);
  cvtT_kernel<<<dim3(1024 * 4096 / 256, 1), 256, 0, stream>>>(w1, w1T, 1024, 4096);
  cvtT_kernel<<<dim3(4096 * 1024 / 256, 1), 256, 0, stream>>>(w2, w2T, 4096, 1024);

  // per-head Q projection: [4096,1024] x [1024,64] per head -> qh[b][h][s][dh]
  gemm_kernel<128, 64, 8, 1, 3><<<dim3(1, 32, 16), 256, 0, stream>>>(
      xb, wqT, wq_b, nullptr, nullptr, qhB, 4096, 64, 1024);

  // flash attention per (b,h), 64 query rows per block
  attn_kernel<<<dim3(32, 32), 128, 0, stream>>>(qhB, ya);

  // fc + bias + residual(x) -> z1 (f32)
  gemm_kernel<128, 128, 4, 2, 1><<<dim3(8, 32, 1), 256, 0, stream>>>(
      ya, fcT, fc_b, x, z1, nullptr, 4096, 1024, 1024);

  // LN1 -> ylnF (f32) + ylnB (bf16)
  ln_kernel<<<4096, 256, 0, stream>>>(z1, ln1g, ln1b, ylnF, ylnB);

  // MLP up: relu(yln @ w1 + b1) -> h1 (bf16)
  gemm_kernel<128, 128, 4, 2, 0><<<dim3(32, 32, 1), 256, 0, stream>>>(
      ylnB, w1T, b1, nullptr, nullptr, h1, 4096, 4096, 1024);

  // MLP down + bias + residual(yln) -> z1 (reuse)
  gemm_kernel<128, 128, 4, 2, 1><<<dim3(8, 32, 1), 256, 0, stream>>>(
      h1, w2T, b2, ylnF, z1, nullptr, 4096, 1024, 4096);

  // LN2 -> final f32 output
  ln_kernel<<<4096, 256, 0, stream>>>(z1, ln2g, ln2b, (float*)d_out, nullptr);
}
